// DeltaNet_31877247271524
// MI455X (gfx1250) — matrix operand order repaired and checked by reading, not compiled, not run on hardware
//
#include <hip/hip_runtime.h>
#include <hip/hip_bf16.h>

typedef __bf16 bf16;
typedef __attribute__((ext_vector_type(16))) __bf16 v16bf;
typedef __attribute__((ext_vector_type(8)))  float  v8f;

// Problem constants (from reference setup_inputs)
constexpr int Bb  = 2;
constexpr int Hh  = 16;
constexpr int Ll  = 4096;
constexpr int Dd  = 128;
constexpr int Cc  = 32;            // chunk size
constexpr int NCH = Ll / Cc;       // 128 chunks
constexpr int DV  = 32;            // value-dim block per workgroup
constexpr int NVB = Dd / DV;       // 4 value blocks -> 128 workgroups total

__device__ __forceinline__ v8f wmma_bf16(v16bf a, v16bf b, v8f c) {
    return __builtin_amdgcn_wmma_f32_16x16x32_bf16(
        /*neg_a=*/false, a, /*neg_b=*/false, b,
        /*c_mod=*/(short)0, c, /*reuse_a=*/false, /*reuse_b=*/false);
}

// ---- gfx1250 async global->LDS copy (ASYNCcnt-tracked) ----
// Low 32 bits of a generic pointer into LDS == wave-relative LDS byte address
// (flat LDS aperture truncates addr[31:0], ISA 10.2).
__device__ __forceinline__ uint32_t lds_addr32(const void* p) {
    return (uint32_t)(size_t)p;
}
__device__ __forceinline__ void async_ld_b128(uint32_t lds_addr, const float* gaddr) {
    asm volatile("global_load_async_to_lds_b128 %0, %1, off"
                 :: "v"(lds_addr), "v"(gaddr) : "memory");
}
__device__ __forceinline__ void async_ld_b32(uint32_t lds_addr, const float* gaddr) {
    asm volatile("global_load_async_to_lds_b32 %0, %1, off"
                 :: "v"(lds_addr), "v"(gaddr) : "memory");
}
__device__ __forceinline__ void wait_async0() {
    asm volatile("s_wait_asynccnt 0x0" ::: "memory");
}

// ---- fragment loaders (wave32; contiguous k order k = 16*(lane/16) + e, same for A and B) ----
// A fragment 16x32 (MxK), element (m,k) = src[(m0+m)*ld + k0+k]
__device__ __forceinline__ v16bf load_a(const bf16* src, int ld, int m0, int k0, int lane) {
    v16bf a;
    const int m  = m0 + (lane & 15);
    const int kb = (lane & 16) ? 16 : 0;
#pragma unroll
    for (int e = 0; e < 16; ++e) {
        const int k = kb + e;   // {0..15} or {16..31}
        a[e] = src[m * ld + k0 + k];
    }
    return a;
}
// transposed A: element (m,k) = src[(k0+k)*ld + m0+m]
__device__ __forceinline__ v16bf load_at(const bf16* src, int ld, int m0, int k0, int lane) {
    v16bf a;
    const int m  = m0 + (lane & 15);
    const int kb = (lane & 16) ? 16 : 0;
#pragma unroll
    for (int e = 0; e < 16; ++e) {
        const int k = kb + e;
        a[e] = src[(k0 + k) * ld + m];
    }
    return a;
}
// A fragment from f32 source (cvt on load)
__device__ __forceinline__ v16bf load_a_f32(const float* src, int ld, int m0, int k0, int lane) {
    v16bf a;
    const int m  = m0 + (lane & 15);
    const int kb = (lane & 16) ? 16 : 0;
#pragma unroll
    for (int e = 0; e < 16; ++e) {
        const int k = kb + e;
        a[e] = (bf16)src[m * ld + k0 + k];
    }
    return a;
}
// B fragment 32x16 (KxN), element (k,n) = src[(k0+k)*ld + n0+n]
__device__ __forceinline__ v16bf load_b(const bf16* src, int ld, int k0, int n0, int lane) {
    v16bf b;
    const int n  = n0 + (lane & 15);
    const int kb = (lane & 16) ? 16 : 0;
#pragma unroll
    for (int e = 0; e < 16; ++e) b[e] = src[(k0 + kb + e) * ld + n];
    return b;
}
// transposed B: element (k,n) = src[(n0+n)*ld + k0+k]   (gives src^T as B)
__device__ __forceinline__ v16bf load_bt(const bf16* src, int ld, int k0, int n0, int lane) {
    v16bf b;
    const int n  = n0 + (lane & 15);
    const int kb = (lane & 16) ? 16 : 0;
#pragma unroll
    for (int e = 0; e < 16; ++e) b[e] = src[n * ld + k0 + kb + e];
    return b;
}
// B fragment from f32 source (cvt on load) — used for state S
__device__ __forceinline__ v16bf load_b_f32(const float* src, int ld, int k0, int n0, int lane) {
    v16bf b;
    const int n  = n0 + (lane & 15);
    const int kb = (lane & 16) ? 16 : 0;
#pragma unroll
    for (int e = 0; e < 16; ++e) b[e] = (bf16)src[(k0 + kb + e) * ld + n];
    return b;
}

__global__ __launch_bounds__(256, 1)
void deltanet_chunk_kernel(const float* __restrict__ q, const float* __restrict__ k,
                           const float* __restrict__ v, const float* __restrict__ beta,
                           float* __restrict__ o, float* __restrict__ Sout) {
    // working LDS (~62 KB)
    __shared__ __align__(16) float S[Dd * DV];      // fp32 running state [dk=128, dv=32]
    __shared__ __align__(16) bf16  qb [Cc * Dd];    // normalized q chunk
    __shared__ __align__(16) bf16  knb[Cc * Dd];    // normalized k chunk
    __shared__ __align__(16) bf16  kbb[Cc * Dd];    // k * beta
    __shared__ __align__(16) bf16  vbb[Cc * DV];    // v * beta (this dv block)
    __shared__ __align__(16) bf16  wb [Cc * Dd];    // -(Tinv @ k_beta)
    __shared__ __align__(16) bf16  uab[Cc * DV];    // u_adj
    __shared__ __align__(16) float Af [Cc * Cc];    // A = kb @ kn^T (fp32)
    __shared__ __align__(16) float Tf [Cc * Cc];    // Tinv (fp32)
    __shared__ __align__(16) bf16  attnb[Cc * Cc];  // tril(q @ kn^T)
    // async staging for next chunk (raw fp32, ~36 KB)
    __shared__ __align__(16) float qst[Cc * Dd];
    __shared__ __align__(16) float kst[Cc * Dd];
    __shared__ __align__(16) float vst[Cc * DV];
    __shared__ __align__(16) float bst[Cc];

    const int bid  = blockIdx.x;
    const int bh   = bid / NVB;
    const int v0   = (bid % NVB) * DV;
    const int t    = threadIdx.x;
    const int lane = t & 31;
    const int wave = t >> 5;

    const float* qg = q    + (size_t)bh * Ll * Dd;
    const float* kg = k    + (size_t)bh * Ll * Dd;
    const float* vg = v    + (size_t)bh * Ll * Dd;
    const float* bg = beta + (size_t)bh * Ll;
    float*       og = o    + (size_t)bh * Ll * Dd;
    float*       Sg = Sout + (size_t)bh * Dd * Dd;

    // issue async staging for a chunk: q/k full rows, v column block, beta
    auto issue_chunk = [&](int ci) {
        const float* qsrc = qg + (size_t)ci * Cc * Dd;
        const float* ksrc = kg + (size_t)ci * Cc * Dd;
#pragma unroll
        for (int p = 0; p < 4; ++p) {
            const int idx = (p * 256 + t) * 4;               // 16B granules over 4096 floats
            async_ld_b128(lds_addr32(qst + idx), qsrc + idx);
            async_ld_b128(lds_addr32(kst + idx), ksrc + idx);
        }
        {
            const int r = t >> 3, c = (t & 7) * 4;           // 32 rows x 32 floats
            async_ld_b128(lds_addr32(vst + r * DV + c),
                          vg + (size_t)(ci * Cc + r) * Dd + v0 + c);
        }
        if (t < Cc)                                          // wave 0 only (uniform)
            async_ld_b32(lds_addr32(bst + t), bg + (size_t)ci * Cc + t);
    };

    for (int i = t; i < Dd * DV; i += 256) S[i] = 0.0f;

    issue_chunk(0);          // prologue: stage chunk 0
    wait_async0();
    __syncthreads();

    for (int ci = 0; ci < NCH; ++ci) {
        // ---------- Stage 1: consume staging: l2norm q/k, beta-scale -> bf16 LDS ----------
        {
            const int row = t >> 3;            // 0..31 (8 threads per row)
            const int seg = t & 7;
            const int c0  = seg * 16;
            const float beta_r = bst[row];

            float qv[16], kv[16];
#pragma unroll
            for (int j = 0; j < 16; ++j) {
                qv[j] = qst[row * Dd + c0 + j];
                kv[j] = kst[row * Dd + c0 + j];
            }
            float ssq_q = 0.f, ssq_k = 0.f;
#pragma unroll
            for (int j = 0; j < 16; ++j) { ssq_q += qv[j]*qv[j]; ssq_k += kv[j]*kv[j]; }
#pragma unroll
            for (int m = 1; m < 8; m <<= 1) {
                ssq_q += __shfl_xor(ssq_q, m, 32);
                ssq_k += __shfl_xor(ssq_k, m, 32);
            }
            const float rq = rsqrtf(ssq_q + 1e-12f);
            const float rk = rsqrtf(ssq_k + 1e-12f);
#pragma unroll
            for (int j = 0; j < 16; ++j) {
                qb [row*Dd + c0 + j] = (bf16)(qv[j] * rq);
                const float kn = kv[j] * rk;
                knb[row*Dd + c0 + j] = (bf16)kn;
                kbb[row*Dd + c0 + j] = (bf16)(kn * beta_r);
            }
            for (int i = t; i < Cc * DV; i += 256)
                vbb[i] = (bf16)(vst[i] * bst[i / DV]);
        }
        __syncthreads();     // staging consumed; bf16 operands published

        // overlap next chunk's HBM traffic with the whole WMMA chain below
        if (ci + 1 < NCH) issue_chunk(ci + 1);

        // ---------- Stage 2: A = kb@kn^T (waves 0-3) ; attn = tril(q@kn^T) (waves 4-7) ----------
        {
            const int mt = (wave >> 1) & 1, nt = wave & 1;
            const bf16* srcA = (wave < 4) ? kbb : qb;
            v8f acc = {};
#pragma unroll
            for (int ks = 0; ks < 4; ++ks) {
                v16bf a = load_a (srcA, Dd, mt*16, ks*32, lane);
                v16bf b = load_bt(knb,  Dd, ks*32, nt*16, lane);
                acc = wmma_bf16(a, b, acc);
            }
            const int nl = lane & 15, mb = (lane >> 4) << 3;
            if (wave < 4) {
#pragma unroll
                for (int r = 0; r < 8; ++r)
                    Af[(mt*16 + mb + r)*Cc + nt*16 + nl] = acc[r];
            } else {
#pragma unroll
                for (int r = 0; r < 8; ++r) {
                    const int m = mt*16 + mb + r, n = nt*16 + nl;
                    attnb[m*Cc + n] = (bf16)((m >= n) ? acc[r] : 0.0f);
                }
            }
        }
        __syncthreads();

        // ---------- Stage 3: Tinv = (I + strict_lower(A))^-1, forward substitution (wave 0) ----------
        if (wave == 0) {
            const int col = lane;
            for (int i = 0; i < Cc; ++i) {
                float s = (i == col) ? 1.0f : 0.0f;
                for (int j = 0; j < i; ++j)
                    s -= Af[i*Cc + j] * Tf[j*Cc + col];
                Tf[i*Cc + col] = s;
            }
        }
        __syncthreads();

        // ---------- Stage 4: wb = -(Tinv @ k_beta)  [32 x 128], 2 tiles/wave ----------
        {
#pragma unroll
            for (int it = 0; it < 2; ++it) {
                const int tt = wave * 2 + it;
                const int mt = tt >> 3, nt = tt & 7;
                v16bf a = load_a_f32(Tf, Cc, mt*16, 0, lane);
                v16bf b = load_b(kbb, Dd, 0, nt*16, lane);
                v8f acc = {};
                acc = wmma_bf16(a, b, acc);
                const int nl = lane & 15, mb = (lane >> 4) << 3;
#pragma unroll
                for (int r = 0; r < 8; ++r)
                    wb[(mt*16 + mb + r)*Dd + nt*16 + nl] = (bf16)(-acc[r]);
            }
        }
        __syncthreads();

        // ---------- Stage 5: u_adj = Tinv@v_beta + wb@S  [32 x 32] (waves 0-3) ----------
        if (wave < 4) {
            const int mt = wave >> 1, nt = wave & 1;
            v8f acc = {};
            {
                v16bf a = load_a_f32(Tf, Cc, mt*16, 0, lane);
                v16bf b = load_b(vbb, DV, 0, nt*16, lane);
                acc = wmma_bf16(a, b, acc);
            }
#pragma unroll
            for (int ks = 0; ks < 4; ++ks) {
                v16bf a = load_a(wb, Dd, mt*16, ks*32, lane);
                v16bf b = load_b_f32(S, DV, ks*32, nt*16, lane);
                acc = wmma_bf16(a, b, acc);
            }
            const int nl = lane & 15, mb = (lane >> 4) << 3;
#pragma unroll
            for (int r = 0; r < 8; ++r)
                uab[(mt*16 + mb + r)*DV + nt*16 + nl] = (bf16)acc[r];
        }
        __syncthreads();

        // ---------- Stage 6: o = q@S + attn@u_adj (waves 0-3) ; dS = kn^T@u_adj (waves 4-7) ----------
        v8f dacc[4] = {};
        if (wave < 4) {
            const int mt = wave >> 1, nt = wave & 1;
            v8f acc = {};
#pragma unroll
            for (int ks = 0; ks < 4; ++ks) {
                v16bf a = load_a(qb, Dd, mt*16, ks*32, lane);
                v16bf b = load_b_f32(S, DV, ks*32, nt*16, lane);
                acc = wmma_bf16(a, b, acc);
            }
            {
                v16bf a = load_a(attnb, Cc, mt*16, 0, lane);
                v16bf b = load_b(uab, DV, 0, nt*16, lane);
                acc = wmma_bf16(a, b, acc);
            }
            const int nl = lane & 15, mb = (lane >> 4) << 3;
#pragma unroll
            for (int r = 0; r < 8; ++r) {
                const int m = mt*16 + mb + r;
                og[(size_t)(ci*Cc + m)*Dd + v0 + nt*16 + nl] = acc[r];
            }
        } else {
            const int w4 = wave - 4;
#pragma unroll
            for (int it = 0; it < 4; ++it) {
                const int tt = w4 * 4 + it;         // 16 tiles of [128 x 32]
                const int mt = tt >> 1, nt = tt & 1;
                v16bf a = load_at(knb, Dd, mt*16, 0, lane);  // kn^T as A
                v16bf b = load_b(uab, DV, 0, nt*16, lane);
                v8f z = {};
                dacc[it] = wmma_bf16(a, b, z);
            }
        }
        __syncthreads();   // all reads of old S complete before update

        // ---------- Stage 7: S += dS (waves 4-7) ----------
        if (wave >= 4) {
            const int w4 = wave - 4;
            const int nl = lane & 15, mb = (lane >> 4) << 3;
#pragma unroll
            for (int it = 0; it < 4; ++it) {
                const int tt = w4 * 4 + it;
                const int mt = tt >> 1, nt = tt & 1;
#pragma unroll
                for (int r = 0; r < 8; ++r) {
                    const int m = mt*16 + mb + r, n = nt*16 + nl;
                    S[m*DV + n] += dacc[it][r];
                }
            }
        }
        // next chunk's staging must be resident before stage 1 reads it
        wait_async0();
        __syncthreads();
    }

    // final state -> d_out tail
    for (int i = t; i < Dd * DV; i += 256) {
        const int r = i / DV, c = i % DV;
        Sg[(size_t)r * Dd + v0 + c] = S[i];
    }
}

extern "C" void kernel_launch(void* const* d_in, const int* in_sizes, int n_in,
                              void* d_out, int out_size, void* d_ws, size_t ws_size,
                              hipStream_t stream) {
    const float* q    = (const float*)d_in[0];
    const float* k    = (const float*)d_in[1];
    const float* v    = (const float*)d_in[2];
    const float* beta = (const float*)d_in[3];
    float* o = (float*)d_out;
    float* S = (float*)d_out + (size_t)Bb * Hh * Ll * Dd;   // outputs concatenated (o, S)

    dim3 grid(Bb * Hh * NVB);   // 128 workgroups
    dim3 block(256);            // 8 waves
    hipLaunchKernelGGL(deltanet_chunk_kernel, grid, block, 0, stream, q, k, v, beta, o, S);
}
